// MultiHeadSelfAttention_85950885527613
// MI455X (gfx1250) — matrix operand order repaired and checked by reading, not compiled, not run on hardware
//
#include <hip/hip_runtime.h>
#include <hip/hip_bf16.h>

typedef __attribute__((ext_vector_type(16))) _Float16 v16h;
typedef __attribute__((ext_vector_type(8)))  _Float16 v8h;
typedef __attribute__((ext_vector_type(8)))  float    v8f;

#define DMODEL   1024
#define NHEADS   16
#define HEADDIM  64
#define SEQT     2048
#define BATCH    4
#define C3       3072
#define MTOT     (BATCH * SEQT)   // 8192

// ---------- WMMA helper ----------
__device__ __forceinline__ v8f wmma_f16(v16h a, v16h b, v8f c) {
  return __builtin_amdgcn_wmma_f32_16x16x32_f16(
      /*neg_a=*/false, a, /*neg_b=*/false, b,
      /*c_mod=*/(short)0, c, /*reuse_a=*/false, /*reuse_b=*/false);
}

// A fragment: 16x32 f16, A row-major [*, lda].
// ISA layout: lane l -> row = l%16; halves 0..7  -> K = 8*(l/16)+h
//                                  halves 8..15 -> K = 16 + 8*(l/16) + (h-8)
__device__ __forceinline__ v16h load_a_frag(const _Float16* __restrict__ A,
                                            int lda, int m0, int k0, int lane) {
  int m  = m0 + (lane & 15);
  int kb = (lane >> 4) << 3;
  const _Float16* p = A + (size_t)m * lda + k0 + kb;
  v8h lo = *reinterpret_cast<const v8h*>(p);
  v8h hi = *reinterpret_cast<const v8h*>(p + 16);
  v16h f;
#pragma unroll
  for (int i = 0; i < 8; ++i) { f[i] = lo[i]; f[i + 8] = hi[i]; }
  return f;
}

// B fragment: 32x16 f16 (KxN) from row-major source Bm[n][k] (NT gemm).
// ISA layout: lane l -> N = l%16; halves 0..7  -> K = 8*(l/16)+h
//                                 halves 8..15 -> K = 16 + 8*(l/16) + (h-8)
__device__ __forceinline__ v16h load_b_frag(const _Float16* __restrict__ Bm,
                                            int ldb, int n0, int k0, int lane) {
  int n  = n0 + (lane & 15);
  int kk = k0 + ((lane >> 4) << 3);
  const _Float16* p = Bm + (size_t)n * ldb + kk;
  v8h lo = *reinterpret_cast<const v8h*>(p);
  v8h hi = *reinterpret_cast<const v8h*>(p + 16);
  v16h f;
#pragma unroll
  for (int i = 0; i < 8; ++i) { f[i] = lo[i]; f[i + 8] = hi[i]; }
  return f;
}

// ---------- fp32 -> f16 convert ----------
__global__ void cvt_f32_f16(const float* __restrict__ src,
                            _Float16* __restrict__ dst, int n) {
  int stride = gridDim.x * blockDim.x;
  for (int i = blockIdx.x * blockDim.x + threadIdx.x; i < n; i += stride)
    dst[i] = (_Float16)src[i];
}

// ---------- QKV projection: qkv[m,n] = sum_k x[m,k]*Wqkv[n,k] + b[n] ----------
// Wave tile 32(M) x 64(N); block = 8 waves (2M x 4N) -> 64 x 256 block tile.
// Scatter epilogue -> Q[B,H,T,D], K[B,H,T,D], Vt[B,H,D,T]  (all f16)
__global__ void qkv_gemm(const _Float16* __restrict__ Xh,
                         const _Float16* __restrict__ Wh,
                         const float* __restrict__ bias,
                         _Float16* __restrict__ Q,
                         _Float16* __restrict__ Kd,
                         _Float16* __restrict__ Vt) {
  int lane = threadIdx.x & 31;
  int wave = threadIdx.x >> 5;
  int wm = wave & 1;
  int wn = wave >> 1;
  int mBase = blockIdx.x * 64 + wm * 32;
  int nBase = blockIdx.y * 256 + wn * 64;

  v8f acc[2][4] = {};
  for (int k0 = 0; k0 < DMODEL; k0 += 32) {
    v16h a0 = load_a_frag(Xh, DMODEL, mBase,      k0, lane);
    v16h a1 = load_a_frag(Xh, DMODEL, mBase + 16, k0, lane);
    v16h b[4];
#pragma unroll
    for (int in = 0; in < 4; ++in)
      b[in] = load_b_frag(Wh, DMODEL, nBase + in * 16, k0, lane);
#pragma unroll
    for (int in = 0; in < 4; ++in) {
      acc[0][in] = wmma_f16(a0, b[in], acc[0][in]);
      acc[1][in] = wmma_f16(a1, b[in], acc[1][in]);
    }
  }

  int half = lane >> 4;
#pragma unroll
  for (int im = 0; im < 2; ++im)
#pragma unroll
    for (int in = 0; in < 4; ++in) {
      int n = nBase + in * 16 + (lane & 15);
      float bv = bias[n];
      int sect = n >> 10;        // 0=Q 1=K 2=V
      int nc = n & 1023;
      int h = nc >> 6, d = nc & 63;
#pragma unroll
      for (int e = 0; e < 8; ++e) {
        int m = mBase + im * 16 + e + (half << 3);
        int b2 = m / SEQT, t = m % SEQT;
        _Float16 hv = (_Float16)(acc[im][in][e] + bv);
        if (sect == 0)
          Q[(((size_t)b2 * NHEADS + h) * SEQT + t) * HEADDIM + d] = hv;
        else if (sect == 1)
          Kd[(((size_t)b2 * NHEADS + h) * SEQT + t) * HEADDIM + d] = hv;
        else
          Vt[(((size_t)b2 * NHEADS + h) * HEADDIM + d) * SEQT + t] = hv;
      }
    }
}

// ---------- causal flash attention ----------
// block = 256 threads (8 waves); wave handles 16 query rows; 32 keys / iter.
__global__ void flash_attn(const _Float16* __restrict__ Q,
                           const _Float16* __restrict__ Kd,
                           const _Float16* __restrict__ Vt,
                           _Float16* __restrict__ Oh) {
  __shared__ _Float16 plds[8][16 * 32];  // per-wave P staging (1 KB each)
  int lane = threadIdx.x & 31;
  int wave = threadIdx.x >> 5;
  int qb = blockIdx.x;
  int h  = blockIdx.y;
  int b  = blockIdx.z;
  int q0 = qb * 128 + wave * 16;

  const _Float16* Qp = Q  + (((size_t)b * NHEADS + h) * SEQT) * HEADDIM;
  const _Float16* Kp = Kd + (((size_t)b * NHEADS + h) * SEQT) * HEADDIM;
  const _Float16* Vp = Vt + (((size_t)b * NHEADS + h) * HEADDIM) * SEQT;

  v16h qf0 = load_a_frag(Qp, HEADDIM, q0, 0,  lane);
  v16h qf1 = load_a_frag(Qp, HEADDIM, q0, 32, lane);

  float mstat[8], lstat[8];
  v8f oacc[4] = {};
#pragma unroll
  for (int e = 0; e < 8; ++e) { mstat[e] = -1e30f; lstat[e] = 0.f; }

  const float scale = 0.125f;  // 1/sqrt(64)
  int half = lane >> 4;
  int col  = lane & 15;
  _Float16* myp = plds[wave];

  for (int k0 = 0; k0 < q0 + 16; k0 += 32) {   // wave-uniform bound
    // S = Q K^T for keys [k0, k0+32): 4 wmma (2 N-tiles x Kdim 64)
    v8f s0 = {}, s1 = {};
    {
      v16h bk;
      bk = load_b_frag(Kp, HEADDIM, k0,      0,  lane); s0 = wmma_f16(qf0, bk, s0);
      bk = load_b_frag(Kp, HEADDIM, k0,      32, lane); s0 = wmma_f16(qf1, bk, s0);
      bk = load_b_frag(Kp, HEADDIM, k0 + 16, 0,  lane); s1 = wmma_f16(qf0, bk, s1);
      bk = load_b_frag(Kp, HEADDIM, k0 + 16, 32, lane); s1 = wmma_f16(qf1, bk, s1);
    }
#pragma unroll
    for (int e = 0; e < 8; ++e) {
      int qrow = q0 + e + (half << 3);
      float v0 = s0[e] * scale;
      float v1 = s1[e] * scale;
      if (k0 + col > qrow)      v0 = -1e30f;   // causal mask
      if (k0 + 16 + col > qrow) v1 = -1e30f;
      // row max across the 16 lanes holding this row (wave32 halves)
      float t = fmaxf(v0, v1);
      for (int off = 8; off >= 1; off >>= 1)
        t = fmaxf(t, __shfl_xor(t, off, 32));
      float mnew  = fmaxf(mstat[e], t);
      float alpha = __expf(mstat[e] - mnew);
      float p0 = __expf(v0 - mnew);
      float p1 = __expf(v1 - mnew);
      float rs = p0 + p1;
      for (int off = 8; off >= 1; off >>= 1)
        rs += __shfl_xor(rs, off, 32);
      lstat[e] = lstat[e] * alpha + rs;
      mstat[e] = mnew;
#pragma unroll
      for (int j = 0; j < 4; ++j) oacc[j][e] *= alpha;
      int prow = e + (half << 3);
      myp[prow * 32 + col]      = (_Float16)p0;
      myp[prow * 32 + 16 + col] = (_Float16)p1;
    }
    asm volatile("s_wait_dscnt 0x0" ::: "memory");
    // reload P in A-fragment layout (16x32)
    v16h pf;
    {
      int row = lane & 15;
      int kb  = half << 3;
      const _Float16* pp = myp + row * 32 + kb;
      v8h lo = *reinterpret_cast<const v8h*>(pp);
      v8h hi = *reinterpret_cast<const v8h*>(pp + 16);
#pragma unroll
      for (int i = 0; i < 8; ++i) { pf[i] = lo[i]; pf[i + 8] = hi[i]; }
    }
    // O += P V : Vt gives contiguous B fragments (ldb = T)
#pragma unroll
    for (int j = 0; j < 4; ++j) {
      v16h bv = load_b_frag(Vp, SEQT, j * 16, k0, lane);
      oacc[j] = wmma_f16(pf, bv, oacc[j]);
    }
  }

  // normalize, store to Oh[B,T,C] (f16)
#pragma unroll
  for (int j = 0; j < 4; ++j)
#pragma unroll
    for (int e = 0; e < 8; ++e) {
      float val = oacc[j][e] / lstat[e];
      int t = q0 + e + (half << 3);
      int c = h * HEADDIM + j * 16 + col;
      Oh[((size_t)b * SEQT + t) * DMODEL + c] = (_Float16)val;
    }
}

// ---------- output projection: out[m,n] = sum_k O[m,k]*Wout[n,k] + b[n] ----------
// Wave tile 32(M) x 64(N); block tile 64 x 256; fp32 epilogue with bias.
__global__ void out_gemm(const _Float16* __restrict__ Ah,
                         const _Float16* __restrict__ Wh,
                         const float* __restrict__ bias,
                         float* __restrict__ out) {
  int lane = threadIdx.x & 31;
  int wave = threadIdx.x >> 5;
  int wm = wave & 1;
  int wn = wave >> 1;
  int mBase = blockIdx.x * 64 + wm * 32;
  int nBase = blockIdx.y * 256 + wn * 64;

  v8f acc[2][4] = {};
  for (int k0 = 0; k0 < DMODEL; k0 += 32) {
    v16h a0 = load_a_frag(Ah, DMODEL, mBase,      k0, lane);
    v16h a1 = load_a_frag(Ah, DMODEL, mBase + 16, k0, lane);
    v16h b[4];
#pragma unroll
    for (int in = 0; in < 4; ++in)
      b[in] = load_b_frag(Wh, DMODEL, nBase + in * 16, k0, lane);
#pragma unroll
    for (int in = 0; in < 4; ++in) {
      acc[0][in] = wmma_f16(a0, b[in], acc[0][in]);
      acc[1][in] = wmma_f16(a1, b[in], acc[1][in]);
    }
  }

  int half = lane >> 4;
#pragma unroll
  for (int im = 0; im < 2; ++im)
#pragma unroll
    for (int in = 0; in < 4; ++in) {
      int n = nBase + in * 16 + (lane & 15);
      float bv = bias[n];
#pragma unroll
      for (int e = 0; e < 8; ++e) {
        int m = mBase + im * 16 + e + (half << 3);
        out[(size_t)m * DMODEL + n] = acc[im][in][e] + bv;
      }
    }
}

extern "C" void kernel_launch(void* const* d_in, const int* in_sizes, int n_in,
                              void* d_out, int out_size, void* d_ws, size_t ws_size,
                              hipStream_t stream) {
  (void)in_sizes; (void)n_in; (void)out_size; (void)ws_size;
  const float* x    = (const float*)d_in[0];
  // d_in[1] = mask: causal mask applied analytically in-kernel
  const float* Wqkv = (const float*)d_in[2];
  const float* bqkv = (const float*)d_in[3];
  const float* Wout = (const float*)d_in[4];
  const float* bout = (const float*)d_in[5];

  char* ws = (char*)d_ws;
  size_t off = 0;
  _Float16* Xh  = (_Float16*)(ws + off); off += (size_t)MTOT * DMODEL * 2;     // 16 MB
  _Float16* Wqh = (_Float16*)(ws + off); off += (size_t)C3 * DMODEL * 2;       //  6 MB
  _Float16* Woh = (_Float16*)(ws + off); off += (size_t)DMODEL * DMODEL * 2;   //  2 MB
  _Float16* Qh  = (_Float16*)(ws + off); off += (size_t)MTOT * DMODEL * 2;     // 16 MB
  _Float16* Kh  = (_Float16*)(ws + off); off += (size_t)MTOT * DMODEL * 2;     // 16 MB
  _Float16* Vth = (_Float16*)(ws + off); off += (size_t)MTOT * DMODEL * 2;     // 16 MB
  _Float16* Oh  = (_Float16*)(ws + off); off += (size_t)MTOT * DMODEL * 2;     // 16 MB

  cvt_f32_f16<<<2048, 256, 0, stream>>>(x,    Xh,  MTOT * DMODEL);
  cvt_f32_f16<<<1024, 256, 0, stream>>>(Wqkv, Wqh, C3 * DMODEL);
  cvt_f32_f16<<<512,  256, 0, stream>>>(Wout, Woh, DMODEL * DMODEL);

  dim3 g1(MTOT / 64, C3 / 256);
  qkv_gemm<<<g1, 256, 0, stream>>>(Xh, Wqh, bqkv, Qh, Kh, Vth);

  dim3 g2(SEQT / 128, NHEADS, BATCH);
  flash_attn<<<g2, 256, 0, stream>>>(Qh, Kh, Vth, Oh);

  dim3 g3(MTOT / 64, DMODEL / 256);
  out_gemm<<<g3, 256, 0, stream>>>(Oh, Woh, bout, (float*)d_out);
}
